// PadeKANLayer_171798692136
// MI455X (gfx1250) — matrix operand order repaired and checked by reading, not compiled, not run on hardware
//
#include <hip/hip_runtime.h>
#include <hip/hip_bf16.h>

typedef __attribute__((ext_vector_type(16))) _Float16 v16h;
typedef __attribute__((ext_vector_type(8)))  float    v8f;

#define IN_F   256
#define OUT_F  256
#define NB     4096
#define SHARP  50.0f
#define PEPS   1e-4f

// fast hardware reciprocal (v_rcp_f32) -- avoids full IEEE div sequence
__device__ __forceinline__ float frcp(float v) { return __builtin_amdgcn_rcpf(v); }

// ---------- prep: per-row l2-norm^2 and product of x ----------
__global__ void pade_prep(const float* __restrict__ x,
                          float* __restrict__ l2o, float* __restrict__ pro) {
  int b = blockIdx.x * blockDim.x + threadIdx.x;
  if (b >= NB) return;
  const float4* row = (const float4*)(x + b * IN_F);
  float s = 0.f, p = 1.f;
  #pragma unroll 4
  for (int i = 0; i < IN_F / 4; ++i) {
    float4 v = row[i];
    s = fmaf(v.x, v.x, fmaf(v.y, v.y, fmaf(v.z, v.z, fmaf(v.w, v.w, s))));
    p *= v.x * v.y * v.z * v.w;
  }
  l2o[b] = s;
  pro[b] = p;
}

// ---------- main: fused base-GEMM + Pade-KAN tile kernel ----------
// One wave = one 16(b) x 16(o) output tile. 4 waves per block (4 o-tiles).
__launch_bounds__(128)
__global__ void pade_kan_main(
    const float* __restrict__ x,     // [NB, IN_F]
    const float* __restrict__ bw,    // [OUT_F, IN_F]
    const float* __restrict__ numw,  // [OUT_F, IN_F, 8]
    const float* __restrict__ denw,  // [OUT_F, IN_F, 8]
    const float* __restrict__ hb,    // [OUT_F, IN_F]
    const float* __restrict__ hw,    // [OUT_F, IN_F]
    const float* __restrict__ l2w,   // [OUT_F, IN_F]
    const float* __restrict__ pw,    // [OUT_F, IN_F]
    const float* __restrict__ l2s,   // [NB]  (ws)
    const float* __restrict__ prs,   // [NB]  (ws)
    float* __restrict__ out)         // [NB, OUT_F]
{
  const int lane = threadIdx.x & 31;
  const int wv   = threadIdx.x >> 5;
  const int half = lane >> 4;   // 0: lanes 0-15, 1: lanes 16-31
  const int lcol = lane & 15;   // N within tile (and A-matrix row M)
  const int b0 = blockIdx.x * 16;
  const int o0 = (blockIdx.y * 4 + wv) * 16;

  const int arow = b0 + lcol;   // x row this lane carries for A-matrix
  const int ocol = o0 + lcol;   // weight row (output feature) this lane carries

  v8f acc = {};   // C/D: VGPR r -> (M = r + 8*half, N = lcol)

  // ===== base GEMM: silu(x) @ base_weight^T, fused into acc via WMMA =====
  for (int ic = 0; ic < IN_F; ic += 32) {
    v16h A, Bm;
    // A 16x32 f16, contiguous k order chosen here (matches B): lanes<16 K=0..15, lanes>=16 K=16..31
    const float* xp0 = x + arow * IN_F + ic + half * 16;
    const float* xp1 = xp0 + 8;
    #pragma unroll
    for (int e = 0; e < 8; ++e) {
      float v0 = xp0[e];
      float v1 = xp1[e];
      A[e]     = (_Float16)(v0 * frcp(1.0f + __expf(-v0)));   // silu
      A[e + 8] = (_Float16)(v1 * frcp(1.0f + __expf(-v1)));
    }
    // B 32x16 f16, same contiguous k order as A: lane carries column N=lcol; lanes<16 K=0..15, >=16 K=16..31
    const float* bp = bw + ocol * IN_F + ic + half * 16;
    #pragma unroll
    for (int e = 0; e < 16; ++e) Bm[e] = (_Float16)bp[e];
    acc = __builtin_amdgcn_wmma_f32_16x16x32_f16(false, A, false, Bm,
                                                 (short)0, acc, false, false);
  }

  // per-lane row scalars for rows M = r + 8*half
  float l2r[8], prr[8];
  #pragma unroll
  for (int r = 0; r < 8; ++r) {
    l2r[r] = l2s[b0 + r + 8 * half];
    prr[r] = prs[b0 + r + 8 * half];
  }

  const float* hbp = hb  + ocol * IN_F;
  const float* hwp = hw  + ocol * IN_F;
  const float* l2p = l2w + ocol * IN_F;
  const float* pwp = pw  + ocol * IN_F;
  const float* nwp = numw + (size_t)ocol * IN_F * 8;
  const float* dwp = denw + (size_t)ocol * IN_F * 8;

  // ===== Pade loop over input features =====
  for (int i = 0; i < IN_F; ++i) {
    float xv = x[arow * IN_F + i];

    // cubic B-spline bases on uniform knots g[j] = (j-3)*0.4 - 1, j=0..11
    float bs[11];
    #pragma unroll
    for (int j = 0; j < 11; ++j) {
      float gl = 0.4f * (float)(j - 3) - 1.0f;
      float gr = 0.4f * (float)(j - 2) - 1.0f;
      bs[j] = (xv >= gl && xv < gr) ? 1.0f : 0.0f;
    }
    #pragma unroll
    for (int k = 1; k <= 3; ++k) {
      float inv = 1.0f / (0.4f * (float)k + 1e-8f);   // compile-time constant
      #pragma unroll
      for (int j = 0; j < 11 - k; ++j) {
        float gj  = 0.4f * (float)(j - 3) - 1.0f;       // g[j]
        float gkr = 0.4f * (float)(j + k - 2) - 1.0f;   // g[j+k+1]
        bs[j] = (xv - gj) * inv * bs[j] + (gkr - xv) * inv * bs[j + 1];
      }
    }

    // K=8 spline dot as padded 16x16x32 WMMA:
    // only lanes<16 / elements 0..7 (K=0..7) are non-zero in A and B.
    v16h A = {}, Bn = {}, Bd = {};
    if (half == 0) {
      const float* np = nwp + i * 8;
      const float* dp = dwp + i * 8;
      #pragma unroll
      for (int e = 0; e < 8; ++e) {
        A[e]  = (_Float16)bs[e];
        Bn[e] = (_Float16)np[e];
        Bd[e] = (_Float16)dp[e];
      }
    }
    v8f z = {};
    v8f nm = __builtin_amdgcn_wmma_f32_16x16x32_f16(false, A, false, Bn,
                                                    (short)0, z, false, false);
    v8f dn = __builtin_amdgcn_wmma_f32_16x16x32_f16(false, A, false, Bd,
                                                    (short)0, z, false, false);

    float hbv = hbp[i], hwv = hwp[i], l2v = l2p[i], pwv = pwp[i];

    #pragma unroll
    for (int r = 0; r < 8; ++r) {
      // row M = r + 8*half; x[b0+M][i] lives in lane M (M < 16)
      float xr = __shfl(xv, r + 8 * half, 32);
      // sigmoid via v_exp + v_rcp
      float sg = frcp(1.0f + __expf(-SHARP * (xr - hbv)));
      float numer = nm[r] + sg * hwv;
      float den = 1.0f + dn[r] + l2r[r] * l2v + prr[r] * pwv;
      // stable softplus: max(v,0) + log(1 + exp(-|v|))
      float sp = fmaxf(den, 0.0f) + __logf(1.0f + __expf(-fabsf(den)));
      // final divide via v_rcp instead of IEEE div sequence
      acc[r] += numer * frcp(sp + PEPS);
    }
  }

  #pragma unroll
  for (int r = 0; r < 8; ++r)
    out[(b0 + r + 8 * half) * OUT_F + o0 + lcol] = acc[r];
}

extern "C" void kernel_launch(void* const* d_in, const int* in_sizes, int n_in,
                              void* d_out, int out_size, void* d_ws, size_t ws_size,
                              hipStream_t stream) {
  const float* x    = (const float*)d_in[0];
  const float* bw   = (const float*)d_in[1];
  const float* numw = (const float*)d_in[2];
  const float* denw = (const float*)d_in[3];
  const float* hb   = (const float*)d_in[4];
  const float* hw   = (const float*)d_in[5];
  const float* l2w  = (const float*)d_in[6];
  const float* pw   = (const float*)d_in[7];
  // d_in[8] = grid; knots are uniform -> recomputed analytically in-kernel.

  float* l2s = (float*)d_ws;       // [NB]
  float* prs = l2s + NB;           // [NB]

  pade_prep<<<dim3(NB / 256), dim3(256), 0, stream>>>(x, l2s, prs);

  // grid: 256 b-tiles x 4 o-groups; block: 4 waves, one 16x16 o-tile each
  pade_kan_main<<<dim3(NB / 16, OUT_F / 64), dim3(128), 0, stream>>>(
      x, bw, numw, denw, hb, hw, l2w, pw, l2s, prs, (float*)d_out);
}
